// OverlapPatchEmbed_46377056862455
// MI455X (gfx1250) — compile-verified, matrix operand order checked by reading, not run on hardware
//
#include <hip/hip_runtime.h>
#include <hip/hip_bf16.h>

// ---------------------------------------------------------------------------
// Fused deformable-conv pipeline for MI455X (gfx1250, wave32, WMMA).
//
//   off2 = softmax(conv3x3(x, w_polar));  out2 = dconv(x, off2, w_dconv)
//   off1 = conv3x3(x, w_dcon_off);        out1 = dconv(x, off1, w_dcon_w)
//   out  = out1 + out2
//
// Shapes: x (8,3,512,512) f32, E=48, K=3x3, contraction C*K = 27 (pad -> 32)
// Strategy: each wave owns a 32-pixel strip; all matrix contractions mapped to
// v_wmma_f32_16x16x32_f16 (27 fits one K step). Both deform einsums accumulate
// into the same f32 WMMA accumulators; both offset convs share one A matrix.
// ---------------------------------------------------------------------------

typedef __attribute__((ext_vector_type(16))) _Float16 v16h;
typedef __attribute__((ext_vector_type(2)))  _Float16 v2h;
typedef __attribute__((ext_vector_type(8)))  float    v8f;

#define HW    512
#define CIN   3
#define EOUT  48
#define KC    27          // CIN * 9 contraction length
#define NSTRIPS (8 * HW * (HW / 32))   // 65536

// 16-bit operand K index for (wave-half hi, half-slot h) per CDNA5 WMMA layout:
// VGPR0..3 (halves 0..7):  lanes 0-15 K=0..7,  lanes 16-31 K=8..15
// VGPR4..7 (halves 8..15): lanes 0-15 K=16..23, lanes 16-31 K=24..31
__device__ __forceinline__ int kOfHalf(int hi, int h) {
    return (h < 8 ? 0 : 16) + hi * 8 + (h & 7);
}

// A fragment for the fused offset convs: rows 0..17 = w_polar, 18..35 = w_dcon_off.
__device__ __forceinline__ v16h loadA_stage1(const float* __restrict__ wp,
                                             const float* __restrict__ wo,
                                             int mtile, int lane) {
    int M = lane & 15, hi = lane >> 4;
    int ch = mtile * 16 + M;
    v16h a;
#pragma unroll
    for (int h = 0; h < 16; ++h) {
        int k = kOfHalf(hi, h);
        float v = 0.0f;
        if (k < KC) {
            if (ch < 18)      v = wp[ch * KC + k];
            else if (ch < 36) v = wo[(ch - 18) * KC + k];
        }
        a[h] = (_Float16)v;
    }
    return a;
}

// A fragment for a deform-conv weight matrix (48 x 27, zero-padded K).
__device__ __forceinline__ v16h loadA_w(const float* __restrict__ w,
                                        int mtile, int lane) {
    int M = lane & 15, hi = lane >> 4;
    int ch = mtile * 16 + M;
    v16h a;
#pragma unroll
    for (int h = 0; h < 16; ++h) {
        int k = kOfHalf(hi, h);
        a[h] = (_Float16)((k < KC) ? w[ch * KC + k] : 0.0f);
    }
    return a;
}

// Build B fragment (32x16 K-major) from LDS tile samp[32 pixels][32 K halves].
__device__ __forceinline__ v16h loadB(const _Float16* __restrict__ samp,
                                      int ntile, int lane) {
    int n = lane & 15, hi = lane >> 4;
    const _Float16* p = samp + (size_t)(ntile * 16 + n) * 32;
    v16h b;
#pragma unroll
    for (int h = 0; h < 16; h += 2) {
        int k = kOfHalf(hi, h);          // pair (h, h+1) -> (k, k+1)
        v2h pr = *(const v2h*)(p + k);   // one ds_load_b32
        b[h]     = pr.x;
        b[h + 1] = pr.y;
    }
    return b;
}

// Bilinear sample 3 channels of one image at (py, px), zero outside (branchless:
// validity folded into corner weights, clamped in-bounds addresses).
__device__ __forceinline__ void bilin3(const float* __restrict__ xb,
                                       float py, float px, float s3[3]) {
    float y0f = floorf(py), x0f = floorf(px);
    float wy = py - y0f, wx = px - x0f;
    int y0 = (int)y0f, x0 = (int)x0f;
    int y1 = y0 + 1,  x1 = x0 + 1;
    float vy0 = (y0 >= 0 && y0 < HW) ? 1.0f : 0.0f;
    float vy1 = (y1 >= 0 && y1 < HW) ? 1.0f : 0.0f;
    float vx0 = (x0 >= 0 && x0 < HW) ? 1.0f : 0.0f;
    float vx1 = (x1 >= 0 && x1 < HW) ? 1.0f : 0.0f;
    int y0c = min(max(y0, 0), HW - 1), y1c = min(max(y1, 0), HW - 1);
    int x0c = min(max(x0, 0), HW - 1), x1c = min(max(x1, 0), HW - 1);
    float w00 = (1.0f - wy) * (1.0f - wx) * vy0 * vx0;
    float w01 = (1.0f - wy) * wx          * vy0 * vx1;
    float w10 = wy          * (1.0f - wx) * vy1 * vx0;
    float w11 = wy          * wx          * vy1 * vx1;
    int i00 = y0c * HW + x0c, i01 = y0c * HW + x1c;
    int i10 = y1c * HW + x0c, i11 = y1c * HW + x1c;
#pragma unroll
    for (int c = 0; c < CIN; ++c) {
        const float* pc = xb + (size_t)c * HW * HW;
        s3[c] = pc[i00] * w00 + pc[i01] * w01 + pc[i10] * w10 + pc[i11] * w11;
    }
}

__device__ __forceinline__ void lds_fence() {
    asm volatile("s_wait_dscnt 0" ::: "memory");
}

extern "C" __global__ void __launch_bounds__(128)
deform_fused_kernel(const float* __restrict__ x,
                    const float* __restrict__ w_polar,
                    const float* __restrict__ w_dconv,
                    const float* __restrict__ w_dcon_off,
                    const float* __restrict__ w_dcon_w,
                    float* __restrict__ out) {
    __shared__ _Float16 lds_samp[4][32][32];   // per-wave sample tile (f16)
    __shared__ float    lds_conv[4][32][36];   // per-wave offset-conv outputs

    const int lane = threadIdx.x & 31;
    const int wid  = threadIdx.x >> 5;
    const int n    = lane & 15;
    const int hi   = lane >> 4;
    _Float16 (*samp)[32] = lds_samp[wid];
    float    (*cvt)[36]  = lds_conv[wid];

    // Invariant A fragments: fused offset-conv weights + both deform weights.
    v16h A1[3], Aw0[3], Aw1[3];
#pragma unroll
    for (int mt = 0; mt < 3; ++mt) {
        A1[mt]  = loadA_stage1(w_polar, w_dcon_off, mt, lane);
        Aw0[mt] = loadA_w(w_dconv, mt, lane);
        Aw1[mt] = loadA_w(w_dcon_w, mt, lane);
    }

    const int totalWaves = gridDim.x * 4;
    for (int s = blockIdx.x * 4 + wid; s < NSTRIPS; s += totalWaves) {
        const int b  = s >> 13;            // / (512*16)
        const int r  = s & 8191;
        const int y  = r >> 4;
        const int x0 = (r & 15) << 5;      // 32-pixel strip start
        const int lx = x0 + lane;          // this lane's pixel column
        const float* xb = x + (size_t)b * CIN * HW * HW;

        // Prefetch next strip's input rows into L2.
        {
            int s2 = s + totalWaves;
            if (s2 < NSTRIPS) {
                int b2 = s2 >> 13, r2 = s2 & 8191;
                const float* pf = x + (size_t)b2 * CIN * HW * HW
                                    + (size_t)(r2 >> 4) * HW + ((r2 & 15) << 5) + lane;
                __builtin_prefetch(pf, 0, 0);
            }
        }

        // ---- Phase 1: im2col of the 3x3 patch (zero pad) into LDS tile ----
#pragma unroll
        for (int k = 0; k < 9; ++k) {
            int yy = y - 1 + k / 3;
            int xx = lx - 1 + k % 3;
            bool ok = (yy >= 0) && (yy < HW) && (xx >= 0) && (xx < HW);
            int yc = min(max(yy, 0), HW - 1);
            int xc = min(max(xx, 0), HW - 1);
#pragma unroll
            for (int c = 0; c < CIN; ++c) {
                float v = ok ? xb[(size_t)c * HW * HW + yc * HW + xc] : 0.0f;
                samp[lane][c * 9 + k] = (_Float16)v;
            }
        }
#pragma unroll
        for (int k = KC; k < 32; ++k) samp[lane][k] = (_Float16)0.0f;
        lds_fence();

        // ---- Offset convs via WMMA: [w_polar; w_dcon_off] (36x27) x patch ----
        {
            v16h B0 = loadB(&samp[0][0], 0, lane);
            v16h B1 = loadB(&samp[0][0], 1, lane);
            v8f acc1[3][2];
#pragma unroll
            for (int mt = 0; mt < 3; ++mt) {
                v8f z = {};
                acc1[mt][0] = __builtin_amdgcn_wmma_f32_16x16x32_f16(
                    false, A1[mt], false, B0, (short)0, z, false, false);
                acc1[mt][1] = __builtin_amdgcn_wmma_f32_16x16x32_f16(
                    false, A1[mt], false, B1, (short)0, z, false, false);
            }
            // Spill the 36 valid channels to LDS (D layout: VGPR r -> M = hi*8+r).
#pragma unroll
            for (int mt = 0; mt < 3; ++mt) {
#pragma unroll
                for (int nt = 0; nt < 2; ++nt) {
                    int px = nt * 16 + n;
#pragma unroll
                    for (int rr = 0; rr < 8; ++rr) {
                        int ch = mt * 16 + hi * 8 + rr;
                        if (ch < 36) cvt[px][ch] = acc1[mt][nt][rr];
                    }
                }
            }
        }
        lds_fence();

        // ---- Two deform passes accumulating into one set of WMMA accs ----
        v8f acc[3][2];
#pragma unroll
        for (int mt = 0; mt < 3; ++mt) {
            v8f z = {};
            acc[mt][0] = z;
            acc[mt][1] = z;
        }

#pragma unroll
        for (int pass = 0; pass < 2; ++pass) {
            float offy[9], offx[9];
            if (pass == 0) {
                // softmax over channels 0..17 of this lane's pixel
                float v[18], m = -3.4e38f;
#pragma unroll
                for (int c = 0; c < 18; ++c) { v[c] = cvt[lane][c]; m = fmaxf(m, v[c]); }
                float sum = 0.0f;
#pragma unroll
                for (int c = 0; c < 18; ++c) { v[c] = __expf(v[c] - m); sum += v[c]; }
                float inv = 1.0f / sum;
#pragma unroll
                for (int k = 0; k < 9; ++k) {
                    offy[k] = v[2 * k] * inv;
                    offx[k] = v[2 * k + 1] * inv;
                }
            } else {
#pragma unroll
                for (int k = 0; k < 9; ++k) {
                    offy[k] = cvt[lane][18 + 2 * k];
                    offx[k] = cvt[lane][18 + 2 * k + 1];
                }
            }

            // Bilinear gather of 27 samples for this lane's pixel -> LDS tile.
#pragma unroll
            for (int k = 0; k < 9; ++k) {
                float py = (float)(y - 1 + k / 3) + offy[k];
                float px = (float)(lx - 1 + k % 3) + offx[k];
                float s3[3];
                bilin3(xb, py, px, s3);
                samp[lane][0 * 9 + k] = (_Float16)s3[0];
                samp[lane][1 * 9 + k] = (_Float16)s3[1];
                samp[lane][2 * 9 + k] = (_Float16)s3[2];
            }
#pragma unroll
            for (int k = KC; k < 32; ++k) samp[lane][k] = (_Float16)0.0f;
            lds_fence();

            v16h B0 = loadB(&samp[0][0], 0, lane);
            v16h B1 = loadB(&samp[0][0], 1, lane);
#pragma unroll
            for (int mt = 0; mt < 3; ++mt) {
                v16h A = (pass == 0) ? Aw0[mt] : Aw1[mt];
                acc[mt][0] = __builtin_amdgcn_wmma_f32_16x16x32_f16(
                    false, A, false, B0, (short)0, acc[mt][0], false, false);
                acc[mt][1] = __builtin_amdgcn_wmma_f32_16x16x32_f16(
                    false, A, false, B1, (short)0, acc[mt][1], false, false);
            }
            lds_fence();   // samp reads done before next pass overwrites
        }

        // ---- Store 48 output channels for the strip (out1 + out2 fused) ----
#pragma unroll
        for (int mt = 0; mt < 3; ++mt) {
#pragma unroll
            for (int nt = 0; nt < 2; ++nt) {
                int px = x0 + nt * 16 + n;
#pragma unroll
                for (int rr = 0; rr < 8; ++rr) {
                    int ch = mt * 16 + hi * 8 + rr;
                    out[(((size_t)b * EOUT + ch) * HW + y) * HW + px] = acc[mt][nt][rr];
                }
            }
        }
    }
}

extern "C" void kernel_launch(void* const* d_in, const int* in_sizes, int n_in,
                              void* d_out, int out_size, void* d_ws, size_t ws_size,
                              hipStream_t stream) {
    const float* x         = (const float*)d_in[0];
    const float* w_polar   = (const float*)d_in[1];
    const float* w_dconv   = (const float*)d_in[2];
    const float* w_dcon_off= (const float*)d_in[3];
    const float* w_dcon_w  = (const float*)d_in[4];
    float* out = (float*)d_out;

    // 65536 strips; 2048 blocks x 4 waves -> each wave loops over 8 strips,
    // amortizing the A-fragment (weight) loads.
    dim3 grid(2048), block(128);
    hipLaunchKernelGGL(deform_fused_kernel, grid, block, 0, stream,
                       x, w_polar, w_dconv, w_dcon_off, w_dcon_w, out);
}
